// QLSTMPlus_65481071403222
// MI455X (gfx1250) — matrix operand order repaired and checked by reading, not compiled, not run on hardware
//
#include <hip/hip_runtime.h>

// ---------------------------------------------------------------------------
// QLSTM persistent kernel for MI455X (gfx1250, wave32, WMMA bf16).
//
// Structure: batch rows are independent -> 16 workgroups, each owning a
// 16-row batch tile, run the full T=512 recurrence with h/c resident in LDS.
// Per step: theta = [x_t | h] @ Wall^T + b via v_wmma_f32_16x16x32_bf16,
// cumprod(cos(theta)) via a chunked LDS scan, then the LSTM elementwise
// update. Weights converted once to bf16 in d_ws (L2-resident, 1 MB).
// ---------------------------------------------------------------------------

#define T_STEPS   512
#define BATCH     256
#define D_INP     256
#define D_HID     256
#define D_COMB    512
#define N_ALL     1024                 // 4 gates * 256
#define TILE_M    16
#define NUM_TILES (BATCH / TILE_M)     // 16 workgroups
#define NTHREADS  256                  // 8 wave32 waves
#define COMB_STRIDE 520                // bf16 elements, padded (row = 1040 B)
#define COS_STRIDE  1028               // f32 elements, padded

typedef __attribute__((ext_vector_type(16))) __bf16 v16bf;
typedef __attribute__((ext_vector_type(8)))  float  v8f;
typedef unsigned int uint4v __attribute__((ext_vector_type(4)));

union AFrag { v16bf v; uint4v u[2]; };

// LDS carve offsets (bytes)
#define OFF_COS    0
#define OFF_CST    (OFF_COS + TILE_M * COS_STRIDE * 4)          // 65792
#define OFF_BIAS   (OFF_CST + TILE_M * D_HID * 4)               // 82176
#define OFF_CHUNK  (OFF_BIAS + N_ALL * 4)                       // 86272
#define OFF_COMB   (OFF_CHUNK + 64 * 4 * 4)                     // 87296 (16B aligned)
#define SMEM_BYTES (OFF_COMB + TILE_M * COMB_STRIDE * 2)        // 103936

__device__ __forceinline__ float sigmoidf_(float x) {
    return 1.0f / (1.0f + __expf(-x));
}

// --- one-time weight fp32 -> bf16 pack: wall[gate][j][k], k contiguous -----
__global__ void qlstm_prep_kernel(const float* __restrict__ Wf,
                                  const float* __restrict__ Wi,
                                  const float* __restrict__ Wg,
                                  const float* __restrict__ Wo,
                                  __bf16* __restrict__ wall) {
    int idx = blockIdx.x * blockDim.x + threadIdx.x;
    if (idx >= 4 * D_HID * D_COMB) return;
    int g = idx >> 17;              // 256*512 = 1<<17
    int r = idx & ((1 << 17) - 1);
    const float* W = (g == 0) ? Wf : (g == 1) ? Wi : (g == 2) ? Wg : Wo;
    wall[idx] = (__bf16)W[r];
}

// --- persistent recurrence kernel ------------------------------------------
__global__ void __launch_bounds__(NTHREADS, 1)
qlstm_main_kernel(const float*  __restrict__ x,
                  const __bf16* __restrict__ wall,
                  const float* __restrict__ bf_, const float* __restrict__ bi_,
                  const float* __restrict__ bg_, const float* __restrict__ bo_,
                  const float* __restrict__ tf_, const float* __restrict__ ti_,
                  const float* __restrict__ tg_, const float* __restrict__ to_,
                  float* __restrict__ out) {
    extern __shared__ char smem[];
    float*  cosLds  = (float*)(smem + OFF_COS);    // [16][COS_STRIDE]
    float*  cstate  = (float*)(smem + OFF_CST);    // [16][256]
    float*  biasLds = (float*)(smem + OFF_BIAS);   // [1024]
    float*  chunkp  = (float*)(smem + OFF_CHUNK);  // [64][4]
    __bf16* comb    = (__bf16*)(smem + OFF_COMB);  // [16][COMB_STRIDE]

    const int tid  = threadIdx.x;
    const int lane = tid & 31;
    const int wave = tid >> 5;
    const int r0   = blockIdx.x * TILE_M;

    const float invtf = 1.0f / tf_[0];
    const float invti = 1.0f / ti_[0];
    const float invtg = 1.0f / tg_[0];
    const float invto = 1.0f / to_[0];

    // init: c = 0, h half of comb = 0, biases to LDS
    for (int row = 0; row < TILE_M; ++row) {
        cstate[row * D_HID + tid] = 0.0f;
        comb[row * COMB_STRIDE + D_INP + tid] = (__bf16)0.0f;
    }
    biasLds[tid]       = bf_[tid];
    biasLds[256 + tid] = bi_[tid];
    biasLds[512 + tid] = bg_[tid];
    biasLds[768 + tid] = bo_[tid];
    __syncthreads();

    // WMMA fragment addressing (CDNA5 layouts)
    const int n0    = wave * 128;            // wave's N slice (8 tiles of 16)
    const int arow  = lane & 15;             // A: lane&15 = M
    const int akoff = (lane >> 4) * 16;      // A: K half-select
    const int bn    = lane & 15;             // B: lane&15 = N
    const int bkoff = (lane >> 4) * 16;      // B: K half-select
    const int mbase = (lane >> 4) * 8;       // C/D: M = i + 8*lane[4]

    for (int t = 0; t < T_STEPS; ++t) {
        // ---- stage x_t (fp32 -> bf16) into first half of comb -------------
        const float* xt = x + ((size_t)t * BATCH + r0) * D_INP;
        for (int row = 0; row < TILE_M; ++row)
            comb[row * COMB_STRIDE + tid] = (__bf16)xt[row * D_INP + tid];
        __syncthreads();

        // ---- theta = comb @ Wall^T + bias (bf16 WMMA, f32 accum) ----------
        v8f acc[8];
        #pragma unroll
        for (int nt = 0; nt < 8; ++nt) {
            float bv = biasLds[n0 + nt * 16 + bn];
            #pragma unroll
            for (int i = 0; i < 8; ++i) acc[nt][i] = bv;
        }
        #pragma unroll 2
        for (int kb = 0; kb < D_COMB / 32; ++kb) {
            AFrag a;
            const __bf16* cp = comb + arow * COMB_STRIDE + kb * 32 + akoff;
            a.u[0] = *(const uint4v*)(cp);
            a.u[1] = *(const uint4v*)(cp + 8);
            const __bf16* wp = wall + (size_t)(n0 + bn) * D_COMB + kb * 32 + bkoff;
            #pragma unroll
            for (int nt = 0; nt < 8; ++nt) {
                v16bf b = *(const v16bf*)(wp + (size_t)nt * 16 * D_COMB);
                acc[nt] = __builtin_amdgcn_wmma_f32_16x16x32_bf16(
                    false, a.v, false, b, (short)0, acc[nt], false, false);
            }
        }
        // ---- cos(theta) -> LDS --------------------------------------------
        #pragma unroll
        for (int nt = 0; nt < 8; ++nt) {
            int n = n0 + nt * 16 + bn;
            #pragma unroll
            for (int i = 0; i < 8; ++i)
                cosLds[(mbase + i) * COS_STRIDE + n] = __cosf(acc[nt][i]);
        }
        __syncthreads();

        // ---- cumprod along hidden dim, per (row, gate) --------------------
        {
            const int pairId = tid >> 2;         // 0..63 = row*4 + gate
            const int chunk  = tid & 3;          // 4 chunks of 64
            const int prow   = pairId >> 2;
            const int pgate  = pairId & 3;
            float* base = cosLds + prow * COS_STRIDE + pgate * 256 + chunk * 64;
            float p = 1.0f;
            #pragma unroll 8
            for (int i = 0; i < 64; ++i) p *= base[i];
            chunkp[pairId * 4 + chunk] = p;
            __syncthreads();
            float run = 1.0f;
            for (int j = 0; j < chunk; ++j) run *= chunkp[pairId * 4 + j];
            #pragma unroll 8
            for (int i = 0; i < 64; ++i) { run *= base[i]; base[i] = run; }
        }
        __syncthreads();

        // ---- gates, state update, outputs ---------------------------------
        const int j = tid;
        for (int row = 0; row < TILE_M; ++row) {
            const float* cr = cosLds + row * COS_STRIDE;
            float f  = sigmoidf_(cr[j]       * invtf);
            float ig = sigmoidf_(cr[256 + j] * invti);
            float g  = tanhf(    cr[512 + j] * invtg);
            float o  = sigmoidf_(cr[768 + j] * invto);
            float c  = f * cstate[row * D_HID + j] + ig * g;
            cstate[row * D_HID + j] = c;
            float h = o * tanhf(c);
            comb[row * COMB_STRIDE + D_INP + j] = (__bf16)h;   // h for t+1
            out[((size_t)t * BATCH + (r0 + row)) * D_HID + j] = h;
            if (t == T_STEPS - 1) {
                size_t hoff = (size_t)T_STEPS * BATCH * D_HID;
                out[hoff + (size_t)(r0 + row) * D_HID + j] = h;
                out[hoff + (size_t)BATCH * D_HID + (size_t)(r0 + row) * D_HID + j] = c;
            }
        }
        __syncthreads();
    }
}

extern "C" void kernel_launch(void* const* d_in, const int* in_sizes, int n_in,
                              void* d_out, int out_size, void* d_ws, size_t ws_size,
                              hipStream_t stream) {
    (void)in_sizes; (void)n_in; (void)out_size; (void)ws_size;
    const float* x  = (const float*)d_in[0];
    const float* Wf = (const float*)d_in[1];
    const float* bf = (const float*)d_in[2];
    const float* tf = (const float*)d_in[3];
    const float* Wi = (const float*)d_in[4];
    const float* bi = (const float*)d_in[5];
    const float* ti = (const float*)d_in[6];
    const float* Wg = (const float*)d_in[7];
    const float* bg = (const float*)d_in[8];
    const float* tg = (const float*)d_in[9];
    const float* Wo = (const float*)d_in[10];
    const float* bo = (const float*)d_in[11];
    const float* to = (const float*)d_in[12];

    __bf16* wall = (__bf16*)d_ws;   // 4*256*512 bf16 = 1 MB (L2-resident)

    const int prep_elems = 4 * D_HID * D_COMB;
    qlstm_prep_kernel<<<(prep_elems + 255) / 256, 256, 0, stream>>>(Wf, Wi, Wg, Wo, wall);

    qlstm_main_kernel<<<NUM_TILES, NTHREADS, SMEM_BYTES, stream>>>(
        x, wall, bf, bi, bg, bo, tf, ti, tg, to, (float*)d_out);
}
